// GATRiskPredictor_15281493639282
// MI455X (gfx1250) — compile-verified, matrix operand order checked by reading, not run on hardware
//
#include <hip/hip_runtime.h>
#include <hip/hip_bf16.h>
#include <math.h>

typedef __bf16 v16bf __attribute__((ext_vector_type(16)));
typedef __bf16 v8bf  __attribute__((ext_vector_type(8)));
typedef float  v8f   __attribute__((ext_vector_type(8)));

#define LEAKY 0.2f
#define EPSV  1e-16f

// ---------------- helpers ----------------
__device__ __forceinline__ __bf16 f2bf(float f) {
    unsigned u = __float_as_uint(f);
    u = u + 0x7FFFu + ((u >> 16) & 1u);           // round-to-nearest-even
    unsigned short s = (unsigned short)(u >> 16);
    union { unsigned short us; __bf16 b; } cv; cv.us = s;
    return cv.b;
}
// order-preserving float<->uint mapping for atomic max
__device__ __forceinline__ unsigned fmap(float f) {
    unsigned u = __float_as_uint(f);
    return (u & 0x80000000u) ? ~u : (u | 0x80000000u);
}
__device__ __forceinline__ float funmap(unsigned u) {
    unsigned v = (u & 0x80000000u) ? (u & 0x7FFFFFFFu) : ~u;
    return __uint_as_float(v);
}

// ---------------- elementwise fp32 -> bf16 ----------------
__global__ void k_f2bf(const float* __restrict__ in, __bf16* __restrict__ out, int n) {
    int i = blockIdx.x * blockDim.x + threadIdx.x;
    if (i < n) out[i] = f2bf(in[i]);
}

// ---------------- W[K,Nc] -> WT[Nc,K] bf16 ----------------
__global__ void k_transpose_bf16(const float* __restrict__ W, __bf16* __restrict__ WT,
                                 int K, int Nc) {
    int i = blockIdx.x * blockDim.x + threadIdx.x;
    if (i < K * Nc) {
        int k = i / Nc, n = i % Nc;
        WT[(size_t)n * K + k] = f2bf(W[i]);
    }
}

// ---------------- WMMA GEMM: C[M,Nc] = A[M,K](bf16) @ WT[Nc,K]^T, fp32 out ----------------
// one wave -> 16x64 output strip (4 wmma tiles), K step 32
__global__ void k_wmma_gemm(const __bf16* __restrict__ A, const __bf16* __restrict__ WT,
                            float* __restrict__ C, int M, int Nc, int K) {
    const int lane  = threadIdx.x & 31;
    const int wave  = threadIdx.x >> 5;
    const int Mtiles = (M + 15) >> 4;
    const int N64    = Nc >> 6;
    int job = blockIdx.x * (blockDim.x >> 5) + wave;
    if (job >= Mtiles * N64) return;               // wave-uniform branch
    const int mtile = job / N64;
    const int ntile = job - mtile * N64;
    const int mbase = mtile << 4;
    const int nbase = ntile << 6;

    const int r     = lane & 15;                   // row (A) / col (B) within tile
    const int khalf = (lane >> 4) << 3;            // lanes 16..31 hold K+8 half

    int rowA = mbase + r; if (rowA >= M) rowA = M - 1;   // clamp keeps EXEC full
    const __bf16* Arow = A + (size_t)rowA * K;
    const __bf16* Bt0  = WT + (size_t)(nbase + r +  0) * K;
    const __bf16* Bt1  = WT + (size_t)(nbase + r + 16) * K;
    const __bf16* Bt2  = WT + (size_t)(nbase + r + 32) * K;
    const __bf16* Bt3  = WT + (size_t)(nbase + r + 48) * K;

    v8f acc0 = {}, acc1 = {}, acc2 = {}, acc3 = {};

    for (int k0 = 0; k0 < K; k0 += 32) {
        const int off0 = k0 + khalf;
        const int off1 = k0 + 16 + khalf;
        __builtin_prefetch(Arow + k0 + 128, 0, 1);   // global_prefetch_b8

        v8bf alo = *(const v8bf*)(Arow + off0);
        v8bf ahi = *(const v8bf*)(Arow + off1);
        v16bf af = __builtin_shufflevector(alo, ahi, 0,1,2,3,4,5,6,7,8,9,10,11,12,13,14,15);

        v8bf b0l = *(const v8bf*)(Bt0 + off0); v8bf b0h = *(const v8bf*)(Bt0 + off1);
        v8bf b1l = *(const v8bf*)(Bt1 + off0); v8bf b1h = *(const v8bf*)(Bt1 + off1);
        v8bf b2l = *(const v8bf*)(Bt2 + off0); v8bf b2h = *(const v8bf*)(Bt2 + off1);
        v8bf b3l = *(const v8bf*)(Bt3 + off0); v8bf b3h = *(const v8bf*)(Bt3 + off1);
        v16bf bf0 = __builtin_shufflevector(b0l, b0h, 0,1,2,3,4,5,6,7,8,9,10,11,12,13,14,15);
        v16bf bf1 = __builtin_shufflevector(b1l, b1h, 0,1,2,3,4,5,6,7,8,9,10,11,12,13,14,15);
        v16bf bf2 = __builtin_shufflevector(b2l, b2h, 0,1,2,3,4,5,6,7,8,9,10,11,12,13,14,15);
        v16bf bf3 = __builtin_shufflevector(b3l, b3h, 0,1,2,3,4,5,6,7,8,9,10,11,12,13,14,15);

        acc0 = __builtin_amdgcn_wmma_f32_16x16x32_bf16(false, af, false, bf0, (short)0, acc0, false, false);
        acc1 = __builtin_amdgcn_wmma_f32_16x16x32_bf16(false, af, false, bf1, (short)0, acc1, false, false);
        acc2 = __builtin_amdgcn_wmma_f32_16x16x32_bf16(false, af, false, bf2, (short)0, acc2, false, false);
        acc3 = __builtin_amdgcn_wmma_f32_16x16x32_bf16(false, af, false, bf3, (short)0, acc3, false, false);
    }

    // C/D layout: VGPR v, lanes 0-15 -> M=v, lanes 16-31 -> M=v+8; N = lane&15
    const int rhalf = (lane >> 4) << 3;
    const int coln  = lane & 15;
    #pragma unroll
    for (int v = 0; v < 8; ++v) {
        int row = mbase + rhalf + v;
        if (row < M) {
            float* cr = C + (size_t)row * Nc + nbase + coln;
            cr[ 0] = acc0[v];
            cr[16] = acc1[v];
            cr[32] = acc2[v];
            cr[48] = acc3[v];
        }
    }
}

// ---------------- attention logits: al_s/al_d[n,h] = <h[n,h,:], a_src/dst[h,:]> ----------------
__global__ void k_attn_logits(const float* __restrict__ h, const float* __restrict__ a_s,
                              const float* __restrict__ a_d, float* __restrict__ al_s,
                              float* __restrict__ al_d, int N, int H, int Cd) {
    int t = blockIdx.x * blockDim.x + threadIdx.x;
    if (t >= N * H) return;
    int n = t / H, hd = t - n * H;
    const float* hp = h + (size_t)n * H * Cd + (size_t)hd * Cd;
    const float* as = a_s + hd * Cd;
    const float* ad = a_d + hd * Cd;
    float ss = 0.f, sd = 0.f;
    for (int c = 0; c < Cd; ++c) { float v = hp[c]; ss += v * as[c]; sd += v * ad[c]; }
    al_s[t] = ss; al_d[t] = sd;
}

// ---------------- init emax/denom/agg ----------------
__global__ void k_init(unsigned* __restrict__ emax, float* __restrict__ denom,
                       float* __restrict__ agg, int nh, int nagg) {
    int i = blockIdx.x * blockDim.x + threadIdx.x;
    if (i < nh) { emax[i] = fmap(-INFINITY); denom[i] = 0.f; }
    if (i < nagg) agg[i] = 0.f;
}

// ---------------- edge pass 1: leaky-relu logits + segment max ----------------
__global__ void k_edge_pass1(const int* __restrict__ ei, const float* __restrict__ al_s,
                             const float* __restrict__ al_d, float* __restrict__ ebuf,
                             unsigned* __restrict__ emax, int E, int N, int H) {
    int t = blockIdx.x * blockDim.x + threadIdx.x;
    int Etot = E + N;
    if (t >= Etot * H) return;
    int e = t / H, hd = t - e * H;
    int s, d;
    if (e < E) { s = ei[e]; d = ei[E + e]; } else { s = e - E; d = s; }
    float x = al_s[s * H + hd] + al_d[d * H + hd];
    x = x > 0.f ? x : LEAKY * x;
    ebuf[t] = x;
    atomicMax(emax + d * H + hd, fmap(x));
}

// ---------------- edge pass 2: exp(e - max) + segment sum ----------------
__global__ void k_edge_pass2(const int* __restrict__ ei, const float* __restrict__ ebuf_in,
                             float* __restrict__ ebuf_out, const unsigned* __restrict__ emax,
                             float* __restrict__ denom, int E, int N, int H) {
    int t = blockIdx.x * blockDim.x + threadIdx.x;
    int Etot = E + N;
    if (t >= Etot * H) return;
    int e = t / H, hd = t - e * H;
    int d = (e < E) ? ei[E + e] : (e - E);
    float m = funmap(emax[d * H + hd]);
    float p = __expf(ebuf_in[t] - m);
    ebuf_out[t] = p;
    atomicAdd(denom + d * H + hd, p);
}

// ---------------- edge pass 3: agg[dst] += h[src] * alpha (one wave per edge) ----------------
__global__ void k_edge_aggregate(const int* __restrict__ ei, const float* __restrict__ h,
                                 const float* __restrict__ ebuf, const float* __restrict__ denom,
                                 float* __restrict__ agg, int E, int N, int H, int Cd) {
    const int lane = threadIdx.x & 31;
    int gw = (blockIdx.x * blockDim.x + threadIdx.x) >> 5;
    int nw = (gridDim.x * blockDim.x) >> 5;
    const int Etot = E + N;
    const int F = H * Cd;
    for (int e = gw; e < Etot; e += nw) {
        int s, d;
        if (e < E) { s = ei[e]; d = ei[E + e]; } else { s = e - E; d = s; }
        float alpha[4];
        for (int hd = 0; hd < H; ++hd)
            alpha[hd] = ebuf[(size_t)e * H + hd] / (denom[d * H + hd] + EPSV);
        const float* hs = h + (size_t)s * F;
        float* od = agg + (size_t)d * F;
        for (int c = lane; c < F; c += 32)
            atomicAdd(od + c, hs[c] * alpha[c / Cd]);
    }
}

// ---------------- epilogues ----------------
__global__ void k_bias_elu_bf16(const float* __restrict__ agg, const float* __restrict__ b,
                                __bf16* __restrict__ out, int N, int F) {
    int t = blockIdx.x * blockDim.x + threadIdx.x;
    if (t >= N * F) return;
    float v = agg[t] + b[t % F];
    v = v > 0.f ? v : expm1f(v);        // ELU(alpha=1)
    out[t] = f2bf(v);
}
__global__ void k_bias_f32(const float* __restrict__ agg, const float* __restrict__ b,
                           float* __restrict__ out, int N, int F) {
    int t = blockIdx.x * blockDim.x + threadIdx.x;
    if (t >= N * F) return;
    out[t] = agg[t] + b[t % F];
}

// ---------------- small fully-connected ----------------
__global__ void k_fc(const float* __restrict__ in, const float* __restrict__ W,
                     const float* __restrict__ b, float* __restrict__ out,
                     int N, int Fin, int Fout, int relu) {
    int t = blockIdx.x * blockDim.x + threadIdx.x;
    if (t >= N * Fout) return;
    int n = t / Fout, j = t - n * Fout;
    const float* x = in + (size_t)n * Fin;
    float s = b[j];
    for (int k = 0; k < Fin; ++k) s += x[k] * W[(size_t)k * Fout + j];
    if (relu) s = fmaxf(s, 0.f);
    out[t] = s;
}

// ---------------- host orchestration ----------------
static inline int cdiv(long long a, int b) { return (int)((a + b - 1) / b); }

static void run_gat_layer(const __bf16* Abf, int K, const float* W, const float* a_s,
                          const float* a_d, int H, int Cd, int N, int E, const int* ei,
                          __bf16* WT, float* h, float* agg, float* al_s, float* al_d,
                          unsigned* emax, float* denom, float* ebuf, hipStream_t stream) {
    const int Nc = H * Cd;
    const int Etot = E + N;
    k_transpose_bf16<<<cdiv((long long)K * Nc, 256), 256, 0, stream>>>(W, WT, K, Nc);
    {
        int jobs = ((N + 15) / 16) * (Nc / 64);
        k_wmma_gemm<<<cdiv(jobs, 8), 256, 0, stream>>>(Abf, WT, h, N, Nc, K);
    }
    k_attn_logits<<<cdiv((long long)N * H, 256), 256, 0, stream>>>(h, a_s, a_d, al_s, al_d, N, H, Cd);
    {
        long long nagg = (long long)N * Nc, nh = (long long)N * H;
        long long mx = nagg > nh ? nagg : nh;
        k_init<<<cdiv(mx, 256), 256, 0, stream>>>(emax, denom, agg, (int)nh, (int)nagg);
    }
    k_edge_pass1<<<cdiv((long long)Etot * H, 256), 256, 0, stream>>>(ei, al_s, al_d, ebuf, emax, E, N, H);
    k_edge_pass2<<<cdiv((long long)Etot * H, 256), 256, 0, stream>>>(ei, ebuf, ebuf, emax, denom, E, N, H);
    k_edge_aggregate<<<1024, 256, 0, stream>>>(ei, h, ebuf, denom, agg, E, N, H, Cd);
}

extern "C" void kernel_launch(void* const* d_in, const int* in_sizes, int n_in,
                              void* d_out, int out_size, void* d_ws, size_t ws_size,
                              hipStream_t stream) {
    const float* x    = (const float*)d_in[0];
    const int*   ei   = (const int*)d_in[1];
    const float* W1   = (const float*)d_in[2];
    const float* a1s  = (const float*)d_in[3];
    const float* a1d  = (const float*)d_in[4];
    const float* b1   = (const float*)d_in[5];
    const float* W2   = (const float*)d_in[6];
    const float* a2s  = (const float*)d_in[7];
    const float* a2d  = (const float*)d_in[8];
    const float* b2   = (const float*)d_in[9];
    const float* W3   = (const float*)d_in[10];
    const float* a3s  = (const float*)d_in[11];
    const float* a3d  = (const float*)d_in[12];
    const float* b3   = (const float*)d_in[13];
    const float* Wc1  = (const float*)d_in[14];
    const float* bc1  = (const float*)d_in[15];
    const float* Wc2  = (const float*)d_in[16];
    const float* bc2  = (const float*)d_in[17];

    const int Fin = 256, Hd = 128, Hh = 4;
    const int N = in_sizes[0] / Fin;
    const int E = in_sizes[1] / 2;
    const int Fhid = Hh * Hd;   // 512

    // workspace carve (256B aligned slices)
    char* wp = (char*)d_ws;
    auto carve = [&](size_t bytes) -> void* {
        void* p = (void*)wp;
        wp += (bytes + 255) & ~(size_t)255;
        return p;
    };
    __bf16*   Abf   = (__bf16*)  carve((size_t)N * Fhid * sizeof(__bf16));
    __bf16*   WT    = (__bf16*)  carve((size_t)Fhid * Fhid * sizeof(__bf16));
    float*    h     = (float*)   carve((size_t)N * Fhid * sizeof(float));
    float*    agg   = (float*)   carve((size_t)N * Fhid * sizeof(float));
    float*    al_s  = (float*)   carve((size_t)N * Hh * sizeof(float));
    float*    al_d  = (float*)   carve((size_t)N * Hh * sizeof(float));
    unsigned* emax  = (unsigned*)carve((size_t)N * Hh * sizeof(unsigned));
    float*    denom = (float*)   carve((size_t)N * Hh * sizeof(float));
    float*    ebuf  = (float*)   carve((size_t)(E + N) * Hh * sizeof(float));
    float*    fc1   = (float*)   carve((size_t)N * (Hd / 2) * sizeof(float));

    float* logits = (float*)d_out;              // [N,2]
    float* emb    = logits + (size_t)N * 2;     // [N,128]

    // layer 1: x -> bf16, GAT(256 -> 4x128), ELU
    k_f2bf<<<cdiv((long long)N * Fin, 256), 256, 0, stream>>>(x, Abf, N * Fin);
    run_gat_layer(Abf, Fin, W1, a1s, a1d, Hh, Hd, N, E, ei,
                  WT, h, agg, al_s, al_d, emax, denom, ebuf, stream);
    k_bias_elu_bf16<<<cdiv((long long)N * Fhid, 256), 256, 0, stream>>>(agg, b1, Abf, N, Fhid);

    // layer 2: GAT(512 -> 4x128), ELU
    run_gat_layer(Abf, Fhid, W2, a2s, a2d, Hh, Hd, N, E, ei,
                  WT, h, agg, al_s, al_d, emax, denom, ebuf, stream);
    k_bias_elu_bf16<<<cdiv((long long)N * Fhid, 256), 256, 0, stream>>>(agg, b2, Abf, N, Fhid);

    // layer 3: GAT(512 -> 1x128), no ELU -> embeddings
    run_gat_layer(Abf, Fhid, W3, a3s, a3d, 1, Hd, N, E, ei,
                  WT, h, agg, al_s, al_d, emax, denom, ebuf, stream);
    k_bias_f32<<<cdiv((long long)N * Hd, 256), 256, 0, stream>>>(agg, b3, emb, N, Hd);

    // classifier: 128 -> 64 (relu) -> 2
    k_fc<<<cdiv((long long)N * (Hd / 2), 256), 256, 0, stream>>>(emb, Wc1, bc1, fc1, N, Hd, Hd / 2, 1);
    k_fc<<<cdiv((long long)N * 2, 256), 256, 0, stream>>>(fc1, Wc2, bc2, logits, N, Hd / 2, 2, 0);
}
